// RGAST_30562987278618
// MI455X (gfx1250) — matrix operand order repaired and checked by reading, not compiled, not run on hardware
//
#include <hip/hip_runtime.h>
#include <hip/hip_bf16.h>
#include <math.h>

// ---------------- problem constants (match reference) ----------------
constexpr int   Nn       = 50000;
constexpr int   Ee       = 1000000;
constexpr int   IN_DIM   = 256;
constexpr int   HID      = 128;
constexpr int   OUTD     = 64;
constexpr int   RREL     = 2;
constexpr float NEG_SLOPE = 0.2f;

typedef __attribute__((ext_vector_type(16))) __bf16 v16bf;
typedef __attribute__((ext_vector_type(8)))  __bf16 v8bf;   // 16 bytes -> global_load_b128
typedef __attribute__((ext_vector_type(8)))  float  v8f;

// ---------------- workspace layout (bytes), lifetime-overlapped ------
constexpr size_t alignup(size_t x) { return (x + 255) & ~size_t(255); }

constexpr size_t XB_OFF   = 0;                                        // bf16 Nn*IN  (later: H1B@+0, H2B after)
constexpr size_t XB_SZ    = alignup((size_t)Nn * IN_DIM * 2);         // 25.6 MB
constexpr size_t H1B_OFF  = XB_OFF;                                   // bf16 Nn*HID (12.8 MB)
constexpr size_t H2B_OFF  = XB_OFF + alignup((size_t)Nn * HID * 2);   // bf16 Nn*OUT (6.4 MB)

constexpr size_t W1B_OFF  = XB_OFF + XB_SZ;                           // bf16 (R,HID,IN)  = W1^T per relation
constexpr size_t W1B_SZ   = alignup((size_t)RREL * IN_DIM * HID * 2);
constexpr size_t W2B_OFF  = W1B_OFF + W1B_SZ;                         // bf16 (R,OUT,HID) = W2^T per relation
constexpr size_t W2B_SZ   = alignup((size_t)RREL * HID * OUTD * 2);
constexpr size_t DW1B_OFF = W2B_OFF + W2B_SZ;                         // bf16 (HID,OUT)   = dec_w1 as-is
constexpr size_t DW1B_SZ  = alignup((size_t)HID * OUTD * 2);
constexpr size_t DW2B_OFF = DW1B_OFF + DW1B_SZ;                       // bf16 (IN,HID)    = dec_w2 as-is
constexpr size_t DW2B_SZ  = alignup((size_t)IN_DIM * HID * 2);

constexpr size_t P_OFF    = DW2B_OFF + DW2B_SZ;                       // f32 R*Nn*HID = 51.2 MB (PROJ1)
constexpr size_t P_SZ     = alignup((size_t)RREL * Nn * HID * 4);
constexpr size_t PROJ2_OFF = P_OFF;                                   // f32 R*Nn*OUT (after layer1 dead)
constexpr size_t T_OFF     = P_OFF + alignup((size_t)RREL * Nn * OUTD * 4); // f32 Nn*HID decoder mid

constexpr size_t AI_OFF   = P_OFF + P_SZ;                             // f32 R*Nn
constexpr size_t AI_SZ    = alignup((size_t)RREL * Nn * 4);
constexpr size_t BJ_OFF   = AI_OFF + AI_SZ;
constexpr size_t BJ_SZ    = AI_SZ;
constexpr size_t AMAX_OFF = BJ_OFF + BJ_SZ;                           // i32 Nn (monotonic-mapped float)
constexpr size_t AMAX_SZ  = alignup((size_t)Nn * 4);
constexpr size_t DEN_OFF  = AMAX_OFF + AMAX_SZ;                       // f32 Nn
constexpr size_t DEN_SZ   = AMAX_SZ;
constexpr size_t EX_OFF   = DEN_OFF + DEN_SZ;                         // f32 Ee (alpha, then exp)
constexpr size_t EX_SZ    = alignup((size_t)Ee * 4);
constexpr size_t H1_OFF   = EX_OFF + EX_SZ;                           // f32 Nn*HID (later: TB bf16 @+0)
constexpr size_t H1_SZ    = alignup((size_t)Nn * HID * 4);
constexpr size_t TB_OFF   = H1_OFF;

// output layout (floats): h2, h3, att1, att2
constexpr size_t OUT_H2   = 0;
constexpr size_t OUT_H3   = (size_t)Nn * OUTD;
constexpr size_t OUT_ATT1 = OUT_H3 + (size_t)Nn * IN_DIM;
constexpr size_t OUT_ATT2 = OUT_ATT1 + (size_t)Ee;

// ---------------- elementwise utility kernels ------------------------
__global__ void rgast_cast_bf16(const float* __restrict__ in, __bf16* __restrict__ out, size_t n) {
  size_t i = (size_t)blockIdx.x * blockDim.x + threadIdx.x;
  if (i < n) out[i] = (__bf16)in[i];
}

// cast + transpose: in is (K,N) row-major f32, out is (N,K) row-major bf16
__global__ void rgast_cast_bf16_t(const float* __restrict__ in, __bf16* __restrict__ out,
                                  int K, int N) {
  int i = blockIdx.x * blockDim.x + threadIdx.x;
  if (i >= K * N) return;
  int k = i / N, n = i - k * N;
  out[(size_t)n * K + k] = (__bf16)in[i];
}

__global__ void rgast_fill_f32(float* __restrict__ p, float v, size_t n) {
  size_t i = (size_t)blockIdx.x * blockDim.x + threadIdx.x;
  if (i < n) p[i] = v;
}

__global__ void rgast_fill_i32(int* __restrict__ p, int v, size_t n) {
  size_t i = (size_t)blockIdx.x * blockDim.x + threadIdx.x;
  if (i < n) p[i] = v;
}

__global__ void rgast_elu_inplace(float* __restrict__ p, size_t n) {
  size_t i = (size_t)blockIdx.x * blockDim.x + threadIdx.x;
  if (i < n) { float x = p[i]; p[i] = x > 0.0f ? x : expm1f(x); }
}

__global__ void rgast_bias_add(float* __restrict__ x, const float* __restrict__ b, size_t n, int D) {
  size_t i = (size_t)blockIdx.x * blockDim.x + threadIdx.x;
  if (i < n) x[i] += b[i % D];
}

// ---------------- WMMA bf16 GEMM: C[M,N] = A[M,K] * Bt^T -------------
// A  : (M,K) row-major bf16, K multiple of 32, rows 16B-aligned
// Bt : (N,K) row-major bf16 (i.e. the mathematical B stored transposed)
// One wave per 16x16 C tile. Per k-step: 4x global_load_b128 + 1x v_wmma.
//
// Fragment layouts (ISA 7.12.2, 16-bit, wave32):
//  A 16x32: lane l<16 row=l, elems 0..7 -> K=k0+e, 8..15 -> K=k0+16+(e-8);
//           lanes 16..31 shift K by +8.  => two contiguous 8-elem runs.
//  B 32x16: lane l<16 col=l, same K mapping as A (elems 0..7 -> K=k0+e,
//           8..15 -> K=k0+16+(e-8); lanes 16..31 shift K by +8).
//  C: VGPR v -> M=v+8*(lane>=16), N=lane&15.
__global__ void rgast_wmma_gemm_bf16(const __bf16* __restrict__ A,
                                     const __bf16* __restrict__ Bt,
                                     float* __restrict__ C,
                                     int M, int N, int K) {
  const int lane = threadIdx.x & 31;
  const int half = lane >> 4;          // 0: lanes 0-15, 1: lanes 16-31
  const int l    = lane & 15;
  const int rowA = blockIdx.x * 16 + l;
  const int rowB = blockIdx.y * 16 + l;

  const v8bf* __restrict__ Ar = (const v8bf*)(A  + (size_t)rowA * K);
  const v8bf* __restrict__ Br = (const v8bf*)(Bt + (size_t)rowB * K);

  v8f acc = {};
  const int steps = K >> 5;            // 32 K per step; 4 v8bf per row per step
  for (int s = 0; s < steps; ++s) {
    const int base = s << 2;
    v8bf alo = Ar[base + half];            // K = 32s + 8*half .. +7      (elems 0..7)
    v8bf ahi = Ar[base + 2 + half];        // K = 32s + 16 + 8*half .. +7 (elems 8..15)
    v8bf blo = Br[base + half];            // K = 32s + 8*half .. +7      (elems 0..7)
    v8bf bhi = Br[base + 2 + half];        // K = 32s + 16 + 8*half .. +7 (elems 8..15)
    v16bf a = __builtin_shufflevector(alo, ahi, 0,1,2,3,4,5,6,7,8,9,10,11,12,13,14,15);
    v16bf b = __builtin_shufflevector(blo, bhi, 0,1,2,3,4,5,6,7,8,9,10,11,12,13,14,15);
    acc = __builtin_amdgcn_wmma_f32_16x16x32_bf16(false, a, false, b,
                                                  (short)0, acc, false, false);
  }

#pragma unroll
  for (int v = 0; v < 8; ++v) {
    int m = blockIdx.x * 16 + v + (half << 3);
    C[(size_t)m * N + (blockIdx.y * 16 + l)] = acc[v];
  }
}

// ---------------- attention scalar / edge kernels --------------------
// ai[i] = proj_row(i) . q ; bj[i] = proj_row(i) . k   (i over R*Nn rows of width O)
__global__ void rgast_node_scalars(const float* __restrict__ proj,
                                   const float* __restrict__ q,
                                   const float* __restrict__ k,
                                   float* __restrict__ ai, float* __restrict__ bj,
                                   int total, int O) {
  int i = blockIdx.x * blockDim.x + threadIdx.x;
  if (i >= total) return;
  const float* row = proj + (size_t)i * O;
  float sq = 0.0f, sk = 0.0f;
  for (int j = 0; j < O; ++j) { float v = row[j]; sq += v * q[j]; sk += v * k[j]; }
  ai[i] = sq; bj[i] = sk;
}

__device__ __forceinline__ int mono_of_f32(float x) {
  int m = __float_as_int(x);
  return m ^ ((m >> 31) & 0x7fffffff);   // order-preserving involution
}

// alpha_e = leaky_relu(ai[t,dst] + bj[t,src]); segment max via atomicMax on mono int
__global__ void rgast_edge_logit_max(const int* __restrict__ src, const int* __restrict__ dst,
                                     const int* __restrict__ et,
                                     const float* __restrict__ ai, const float* __restrict__ bj,
                                     float* __restrict__ alpha, int* __restrict__ amax, int E_) {
  int e = blockIdx.x * blockDim.x + threadIdx.x;
  if (e >= E_) return;
  int s = src[e], d = dst[e], t = et[e];
  float x = ai[t * Nn + d] + bj[t * Nn + s];
  x = x > 0.0f ? x : NEG_SLOPE * x;
  alpha[e] = x;
  atomicMax(&amax[d], mono_of_f32(x));
}

// ex_e = exp(alpha_e - amax[dst]); denom[dst] += ex_e  (alpha buffer overwritten)
__global__ void rgast_edge_exp(const int* __restrict__ dst, float* __restrict__ aex,
                               const int* __restrict__ amax, float* __restrict__ denom, int E_) {
  int e = blockIdx.x * blockDim.x + threadIdx.x;
  if (e >= E_) return;
  int d = dst[e];
  int m = amax[d]; m ^= (m >> 31) & 0x7fffffff;
  float ex = expf(aex[e] - __int_as_float(m));
  aex[e] = ex;
  atomicAdd(&denom[d], ex);
}

// att = ex/(denom+1e-16); out[dst,:] += att * proj[type,src,:]. block = (256/O) edges x O feats
__global__ void rgast_edge_aggregate(const int* __restrict__ src, const int* __restrict__ dst,
                                     const int* __restrict__ et,
                                     const float* __restrict__ ex, const float* __restrict__ denom,
                                     const float* __restrict__ proj,
                                     float* __restrict__ out, float* __restrict__ att,
                                     int E_, int O) {
  int epb = blockDim.x / O;
  int el  = threadIdx.x / O;
  int f   = threadIdx.x - el * O;
  int e   = blockIdx.x * epb + el;
  if (e >= E_) return;
  int d = dst[e];
  float a = ex[e] / (denom[d] + 1e-16f);
  if (f == 0) att[e] = a;
  int s = src[e], t = et[e];
  float v = proj[((size_t)t * Nn + s) * O + f];
  atomicAdd(&out[(size_t)d * O + f], a * v);
}

// ---------------- host orchestration ---------------------------------
static inline unsigned cdiv(size_t a, unsigned b) { return (unsigned)((a + b - 1) / b); }

static void launch_gemm(const __bf16* A, const __bf16* Bt, float* C,
                        int M, int N, int K, hipStream_t s) {
  dim3 grid(M / 16, N / 16);
  rgast_wmma_gemm_bf16<<<grid, dim3(32), 0, s>>>(A, Bt, C, M, N, K);
}

extern "C" void kernel_launch(void* const* d_in, const int* in_sizes, int n_in,
                              void* d_out, int out_size, void* d_ws, size_t ws_size,
                              hipStream_t stream) {
  (void)in_sizes; (void)n_in; (void)out_size; (void)ws_size;
  const float* features = (const float*)d_in[0];
  const int*   eidx     = (const int*)d_in[1];
  const int*   etype    = (const int*)d_in[2];
  const float* W1       = (const float*)d_in[3];
  const float* q1       = (const float*)d_in[4];
  const float* k1       = (const float*)d_in[5];
  const float* W2       = (const float*)d_in[6];
  const float* q2       = (const float*)d_in[7];
  const float* k2       = (const float*)d_in[8];
  const float* dw1      = (const float*)d_in[9];
  const float* db1      = (const float*)d_in[10];
  const float* dw2      = (const float*)d_in[11];
  const float* db2      = (const float*)d_in[12];

  const int* src = eidx;          // edge_index[0]
  const int* dst = eidx + Ee;     // edge_index[1]

  char* ws = (char*)d_ws;
  __bf16* XB   = (__bf16*)(ws + XB_OFF);
  __bf16* W1T  = (__bf16*)(ws + W1B_OFF);   // (R, HID, IN)
  __bf16* W2T  = (__bf16*)(ws + W2B_OFF);   // (R, OUT, HID)
  __bf16* DW1B = (__bf16*)(ws + DW1B_OFF);  // (HID, OUT)
  __bf16* DW2B = (__bf16*)(ws + DW2B_OFF);  // (IN, HID)
  float*  PROJ1 = (float*)(ws + P_OFF);
  float*  PROJ2 = (float*)(ws + PROJ2_OFF);
  float*  Tmid  = (float*)(ws + T_OFF);
  float*  AI    = (float*)(ws + AI_OFF);
  float*  BJ    = (float*)(ws + BJ_OFF);
  int*    AMAX  = (int*)  (ws + AMAX_OFF);
  float*  DEN   = (float*)(ws + DEN_OFF);
  float*  EX    = (float*)(ws + EX_OFF);
  float*  H1    = (float*)(ws + H1_OFF);
  __bf16* H1B   = (__bf16*)(ws + H1B_OFF);
  __bf16* H2B   = (__bf16*)(ws + H2B_OFF);
  __bf16* TB    = (__bf16*)(ws + TB_OFF);

  float* out  = (float*)d_out;
  float* oH2  = out + OUT_H2;
  float* oH3  = out + OUT_H3;
  float* oA1  = out + OUT_ATT1;
  float* oA2  = out + OUT_ATT2;

  const unsigned TB256 = 256;

  // ---- stage weights/activations to bf16 (weights transposed to (N,K)) ----
  rgast_cast_bf16<<<cdiv((size_t)Nn * IN_DIM, TB256), TB256, 0, stream>>>(features, XB, (size_t)Nn * IN_DIM);
  for (int r = 0; r < RREL; ++r) {
    rgast_cast_bf16_t<<<cdiv((size_t)IN_DIM * HID, TB256), TB256, 0, stream>>>(
        W1 + (size_t)r * IN_DIM * HID, W1T + (size_t)r * HID * IN_DIM, IN_DIM, HID);
    rgast_cast_bf16_t<<<cdiv((size_t)HID * OUTD, TB256), TB256, 0, stream>>>(
        W2 + (size_t)r * HID * OUTD, W2T + (size_t)r * OUTD * HID, HID, OUTD);
  }
  rgast_cast_bf16<<<cdiv((size_t)HID * OUTD, TB256), TB256, 0, stream>>>(dw1, DW1B, (size_t)HID * OUTD);
  rgast_cast_bf16<<<cdiv((size_t)IN_DIM * HID, TB256), TB256, 0, stream>>>(dw2, DW2B, (size_t)IN_DIM * HID);

  // ================= layer 1 =================
  for (int r = 0; r < RREL; ++r)
    launch_gemm(XB, W1T + (size_t)r * HID * IN_DIM, PROJ1 + (size_t)r * Nn * HID,
                Nn, HID, IN_DIM, stream);

  rgast_node_scalars<<<cdiv((size_t)RREL * Nn, TB256), TB256, 0, stream>>>(PROJ1, q1, k1, AI, BJ, RREL * Nn, HID);
  rgast_fill_i32<<<cdiv((size_t)Nn, TB256), TB256, 0, stream>>>(AMAX, (int)0x80000000, (size_t)Nn);
  rgast_fill_f32<<<cdiv((size_t)Nn, TB256), TB256, 0, stream>>>(DEN, 0.0f, (size_t)Nn);

  rgast_edge_logit_max<<<cdiv((size_t)Ee, TB256), TB256, 0, stream>>>(src, dst, etype, AI, BJ, EX, AMAX, Ee);
  rgast_edge_exp<<<cdiv((size_t)Ee, TB256), TB256, 0, stream>>>(dst, EX, AMAX, DEN, Ee);

  rgast_fill_f32<<<cdiv((size_t)Nn * HID, TB256), TB256, 0, stream>>>(H1, 0.0f, (size_t)Nn * HID);
  {
    unsigned epb = TB256 / HID;                 // 2 edges per block
    rgast_edge_aggregate<<<cdiv((size_t)Ee, epb), TB256, 0, stream>>>(src, dst, etype, EX, DEN, PROJ1, H1, oA1, Ee, HID);
  }
  rgast_elu_inplace<<<cdiv((size_t)Nn * HID, TB256), TB256, 0, stream>>>(H1, (size_t)Nn * HID);
  rgast_cast_bf16<<<cdiv((size_t)Nn * HID, TB256), TB256, 0, stream>>>(H1, H1B, (size_t)Nn * HID);  // H1B overlays dead XB

  // ================= layer 2 =================
  for (int r = 0; r < RREL; ++r)
    launch_gemm(H1B, W2T + (size_t)r * OUTD * HID, PROJ2 + (size_t)r * Nn * OUTD,
                Nn, OUTD, HID, stream);

  rgast_node_scalars<<<cdiv((size_t)RREL * Nn, TB256), TB256, 0, stream>>>(PROJ2, q2, k2, AI, BJ, RREL * Nn, OUTD);
  rgast_fill_i32<<<cdiv((size_t)Nn, TB256), TB256, 0, stream>>>(AMAX, (int)0x80000000, (size_t)Nn);
  rgast_fill_f32<<<cdiv((size_t)Nn, TB256), TB256, 0, stream>>>(DEN, 0.0f, (size_t)Nn);

  rgast_edge_logit_max<<<cdiv((size_t)Ee, TB256), TB256, 0, stream>>>(src, dst, etype, AI, BJ, EX, AMAX, Ee);
  rgast_edge_exp<<<cdiv((size_t)Ee, TB256), TB256, 0, stream>>>(dst, EX, AMAX, DEN, Ee);

  rgast_fill_f32<<<cdiv((size_t)Nn * OUTD, TB256), TB256, 0, stream>>>(oH2, 0.0f, (size_t)Nn * OUTD);
  {
    unsigned epb = TB256 / OUTD;                // 4 edges per block
    rgast_edge_aggregate<<<cdiv((size_t)Ee, epb), TB256, 0, stream>>>(src, dst, etype, EX, DEN, PROJ2, oH2, oA2, Ee, OUTD);
  }
  rgast_elu_inplace<<<cdiv((size_t)Nn * OUTD, TB256), TB256, 0, stream>>>(oH2, (size_t)Nn * OUTD);
  rgast_cast_bf16<<<cdiv((size_t)Nn * OUTD, TB256), TB256, 0, stream>>>(oH2, H2B, (size_t)Nn * OUTD);

  // ================= decoder =================
  // t = h2 @ dec_w1^T + b1   (dec_w1 is (HID,OUT) row-major == Bt layout directly)
  launch_gemm(H2B, DW1B, Tmid, Nn, HID, OUTD, stream);
  rgast_bias_add<<<cdiv((size_t)Nn * HID, TB256), TB256, 0, stream>>>(Tmid, db1, (size_t)Nn * HID, HID);
  rgast_cast_bf16<<<cdiv((size_t)Nn * HID, TB256), TB256, 0, stream>>>(Tmid, TB, (size_t)Nn * HID);  // TB overlays dead H1

  // h3 = t @ dec_w2^T + b2   (dec_w2 is (IN,HID) row-major == Bt layout directly)
  launch_gemm(TB, DW2B, oH3, Nn, IN_DIM, HID, stream);
  rgast_bias_add<<<cdiv((size_t)Nn * IN_DIM, TB256), TB256, 0, stream>>>(oH3, db2, (size_t)Nn * IN_DIM, IN_DIM);
}
